// Int8WeightLinear_2259152798357
// MI455X (gfx1250) — 8-bit matrix operand order repaired and checked by reading, compiled, not run on hardware
//
#include <hip/hip_runtime.h>

typedef __attribute__((ext_vector_type(2))) int v2i;
typedef __attribute__((ext_vector_type(4))) int v4i;
typedef __attribute__((ext_vector_type(8))) int v8i;

#define MROWS 8192
#define NOUT  4096
#define KDIM  4096
#define QMAXF 127.0f
#define EPSF  1e-8f

#define BM 128
#define BN 128
#define BK 64
#define LDST 80
#define KT (KDIM / BK)

#if defined(__has_builtin)
#if __has_builtin(__builtin_amdgcn_global_load_async_to_lds_b128)
#define HAS_ASYNC_LDS 1
#endif
#endif
#ifndef HAS_ASYNC_LDS
#define HAS_ASYNC_LDS 0
#endif

typedef __attribute__((address_space(1))) v4i* gv4p;
typedef __attribute__((address_space(3))) v4i* lv4p;

__device__ __forceinline__ gv4p to_global_v4(const void* p) {
  return (gv4p)(unsigned long long)p;
}
__device__ __forceinline__ lv4p to_lds_v4(const void* p) {
  return (lv4p)(unsigned)(unsigned long long)p;
}

__device__ __forceinline__ void wait_async0() {
#if defined(__has_builtin) && __has_builtin(__builtin_amdgcn_s_wait_asynccnt)
  __builtin_amdgcn_s_wait_asynccnt(0);
#else
  asm volatile("s_wait_asynccnt 0x0" ::: "memory");
#endif
}

__global__ __launch_bounds__(256) void quant_rows_kernel(
    const float* __restrict__ src, signed char* __restrict__ q,
    float* __restrict__ scale_out, int fp16_round_scale) {
  const int row = blockIdx.x;
  const float* r = src + (size_t)row * KDIM;
  const int t = threadIdx.x;

  float m = 0.0f;
  for (int j = t; j < KDIM; j += 256) m = fmaxf(m, fabsf(r[j]));
#pragma unroll
  for (int off = 16; off > 0; off >>= 1)
    m = fmaxf(m, __shfl_xor(m, off, 32));

  __shared__ float red[8];
  __shared__ float sS;
  if ((t & 31) == 0) red[t >> 5] = m;
  __syncthreads();
  if (t == 0) {
    float mm = red[0];
#pragma unroll
    for (int i = 1; i < 8; ++i) mm = fmaxf(mm, red[i]);
    float s = fmaxf(mm, EPSF) / QMAXF;
    sS = s;
    scale_out[row] = fp16_round_scale ? (float)(_Float16)s : s;
  }
  __syncthreads();

  const float s = sS;
  signed char* qr = q + (size_t)row * KDIM;
  for (int j = t; j < KDIM; j += 256) {
    float v = rintf(r[j] / s);
    v = fminf(fmaxf(v, -QMAXF), QMAXF);
    qr[j] = (signed char)v;
  }
}

__global__ __launch_bounds__(256) void int8_gemm_wmma_kernel(
    const signed char* __restrict__ xq, const signed char* __restrict__ wq,
    const float* __restrict__ xsc, const float* __restrict__ wsc,
    const float* __restrict__ bias, float* __restrict__ out) {
  __shared__ __align__(16) signed char sA[2][BM * LDST];
  __shared__ __align__(16) signed char sB[2][BN * LDST];

  const int t    = threadIdx.x;
  const int lane = t & 31;
  const int w    = t >> 5;
  const int wm   = w & 3;
  const int wn   = w >> 2;
  const int l16  = lane & 15;
  const int lh   = lane >> 4;
  const int bm   = blockIdx.y;
  const int bn   = blockIdx.x;

  const signed char* gA = xq + (size_t)bm * BM * KDIM;
  const signed char* gB = wq + (size_t)bn * BN * KDIM;

  const int id0 = t, id1 = t + 256;
  const int r0 = id0 >> 2, c0 = (id0 & 3) << 4;
  const int r1 = id1 >> 2, c1 = (id1 & 3) << 4;

#if HAS_ASYNC_LDS
  auto issue_copy = [&](int kt, int buf) {
    const signed char* pa = gA + kt * BK;
    const signed char* pb = gB + kt * BK;
    __builtin_amdgcn_global_load_async_to_lds_b128(
        to_global_v4(pa + (size_t)r0 * KDIM + c0),
        to_lds_v4(sA[buf] + r0 * LDST + c0), 0, 0);
    __builtin_amdgcn_global_load_async_to_lds_b128(
        to_global_v4(pa + (size_t)r1 * KDIM + c1),
        to_lds_v4(sA[buf] + r1 * LDST + c1), 0, 0);
    __builtin_amdgcn_global_load_async_to_lds_b128(
        to_global_v4(pb + (size_t)r0 * KDIM + c0),
        to_lds_v4(sB[buf] + r0 * LDST + c0), 0, 0);
    __builtin_amdgcn_global_load_async_to_lds_b128(
        to_global_v4(pb + (size_t)r1 * KDIM + c1),
        to_lds_v4(sB[buf] + r1 * LDST + c1), 0, 0);
  };
#else
  v4i ra0, ra1, rb0, rb1;
  auto g2r = [&](int kt) {
    const signed char* pa = gA + kt * BK;
    const signed char* pb = gB + kt * BK;
    ra0 = *(const v4i*)(pa + (size_t)r0 * KDIM + c0);
    ra1 = *(const v4i*)(pa + (size_t)r1 * KDIM + c1);
    rb0 = *(const v4i*)(pb + (size_t)r0 * KDIM + c0);
    rb1 = *(const v4i*)(pb + (size_t)r1 * KDIM + c1);
  };
  auto r2s = [&](int buf) {
    *(v4i*)(sA[buf] + r0 * LDST + c0) = ra0;
    *(v4i*)(sA[buf] + r1 * LDST + c1) = ra1;
    *(v4i*)(sB[buf] + r0 * LDST + c0) = rb0;
    *(v4i*)(sB[buf] + r1 * LDST + c1) = rb1;
  };
#endif

  v8i acc[2][4];
  const v8i zero8 = {0, 0, 0, 0, 0, 0, 0, 0};
#pragma unroll
  for (int i = 0; i < 2; ++i)
#pragma unroll
    for (int j = 0; j < 4; ++j) acc[i][j] = zero8;

#if HAS_ASYNC_LDS
  issue_copy(0, 0);
  wait_async0();
#else
  g2r(0);
  r2s(0);
#endif
  __syncthreads();

  for (int kt = 0; kt < KT; ++kt) {
    const int cur = kt & 1;

#if HAS_ASYNC_LDS
    if (kt + 1 < KT) issue_copy(kt + 1, cur ^ 1);
#else
    if (kt + 1 < KT) g2r(kt + 1);
#endif

    v8i af[2];
#pragma unroll
    for (int ti = 0; ti < 2; ++ti) {
      const signed char* base =
          sA[cur] + (wm * 32 + ti * 16 + l16) * LDST + 16 * lh;
#pragma unroll
      for (int c = 0; c < 4; ++c) {
        v2i d = *(const v2i*)(base + 8 * (c & 1) + 32 * (c >> 1));
        af[ti][2 * c]     = d.x;
        af[ti][2 * c + 1] = d.y;
      }
    }
    v8i bf[4];
#pragma unroll
    for (int tj = 0; tj < 4; ++tj) {
      const signed char* base =
          sB[cur] + (wn * 64 + tj * 16 + l16) * LDST + 16 * lh;
      v4i lo = *(const v4i*)(base);
      v4i hi = *(const v4i*)(base + 32);
      v8i b;
      b[0] = lo.x; b[1] = lo.y; b[2] = lo.z; b[3] = lo.w;
      b[4] = hi.x; b[5] = hi.y; b[6] = hi.z; b[7] = hi.w;
      bf[tj] = b;
    }

#pragma unroll
    for (int ti = 0; ti < 2; ++ti)
#pragma unroll
      for (int tj = 0; tj < 4; ++tj)
        acc[ti][tj] = __builtin_amdgcn_wmma_i32_16x16x64_iu8(
             true, af[ti],  true, bf[tj], acc[ti][tj],
             false,  false);

#if HAS_ASYNC_LDS
    if (kt + 1 < KT) wait_async0();
#else
    if (kt + 1 < KT) r2s((kt + 1) & 1);
#endif
    __syncthreads();
  }

#pragma unroll
  for (int tj = 0; tj < 4; ++tj) {
    const int n = bn * BN + wn * 64 + tj * 16 + l16;
    const float wsn = wsc[n];
    const float bb = (float)(_Float16)bias[n];
#pragma unroll
    for (int ti = 0; ti < 2; ++ti) {
      const int mbase = bm * BM + wm * 32 + ti * 16 + 8 * lh;
#pragma unroll
      for (int v = 0; v < 8; ++v) {
        const int m = mbase + v;
        out[(size_t)m * NOUT + n] =
            (float)acc[ti][tj][v] * xsc[m] * wsn + bb;
      }
    }
  }
}

extern "C" void kernel_launch(void* const* d_in, const int* in_sizes, int n_in,
                              void* d_out, int out_size, void* d_ws,
                              size_t ws_size, hipStream_t stream) {
  const float* x      = (const float*)d_in[0];
  const float* weight = (const float*)d_in[1];
  const float* bias   = (const float*)d_in[2];
  float* out          = (float*)d_out;

  char* ws = (char*)d_ws;
  signed char* wq  = (signed char*)ws;
  signed char* xq  = (signed char*)(ws + (size_t)16777216);
  float*       wsc = (float*)(ws + (size_t)16777216 + 33554432);
  float*       xsc = (float*)(ws + (size_t)16777216 + 33554432 + 16384);

  quant_rows_kernel<<<NOUT, 256, 0, stream>>>(weight, wq, wsc, 1);
  quant_rows_kernel<<<MROWS, 256, 0, stream>>>(x, xq, xsc, 0);

  dim3 grid(NOUT / BN, MROWS / BM);
  int8_gemm_wmma_kernel<<<grid, 256, 0, stream>>>(xq, wq, xsc, wsc, bias, out);
}
